// ScatteringResidualConvBlock_1683627180751
// MI455X (gfx1250) — matrix operand order repaired and checked by reading, not compiled, not run on hardware
//
#include <hip/hip_runtime.h>
#include <hip/hip_bf16.h>
#include <math.h>

typedef _Float16 h16;
typedef __attribute__((ext_vector_type(16))) _Float16 v16h;
typedef __attribute__((ext_vector_type(8)))  _Float16 v8h;
typedef __attribute__((ext_vector_type(4)))  _Float16 v4h;
typedef __attribute__((ext_vector_type(8)))  float    v8f;

#define Hc   26
#define Wc   26
#define HWc  (Hc*Wc)        // 676
#define Hp   28
#define Wp   28
#define HWp  (Hp*Wp)        // 784
#define Bc   256
#define DDc  128
#define CHc  256
#define Lc   624
#define NPIX (Bc*HWc)       // 173056
#define NLc  4

// ---------------- small utility kernels ----------------

__global__ void cvt_f16_k(const float* __restrict__ src, h16* __restrict__ dst, int n) {
    int i = blockIdx.x * 256 + threadIdx.x;
    if (i < n) dst[i] = (h16)src[i];
}

__global__ void zero_f16_k(h16* __restrict__ dst, int n8) {  // n8 = count of v8h chunks
    int i = blockIdx.x * 256 + threadIdx.x;
    if (i < n8) { v8h z = {}; *(v8h*)(dst + (long)i * 8) = z; }
}

__global__ void init_grid_k(float* __restrict__ Y, const float* __restrict__ P, int n) {
    int i = blockIdx.x * 256 + threadIdx.x;
    if (i < n) Y[i] = P[i & (DDc - 1)];
}

__global__ void scatter_k(const float* __restrict__ X, const int* __restrict__ coords,
                          float* __restrict__ Y, int n) {
    int i = blockIdx.x * 256 + threadIdx.x;
    if (i >= n) return;
    int c  = i & (DDc - 1);
    int bl = i >> 7;
    int l  = bl % Lc;
    int b  = bl / Lc;
    int row = coords[2 * l + 0];
    int col = coords[2 * l + 1];
    Y[((long)(b * HWc + row * Wc + col)) * DDc + c] = X[i];
}

__global__ void gather_k(const float* __restrict__ Y, const int* __restrict__ coords,
                         float* __restrict__ Z, int n) {
    int i = blockIdx.x * 256 + threadIdx.x;
    if (i >= n) return;
    int c  = i & (DDc - 1);
    int bl = i >> 7;
    int l  = bl % Lc;
    int b  = bl / Lc;
    int row = coords[2 * l + 0];
    int col = coords[2 * l + 1];
    Z[i] = Y[((long)(b * HWc + row * Wc + col)) * DDc + c];
}

// ---------------- LayerNorm: Y(f32) -> padded Yn(f16), one wave per pixel ----------------

__global__ __launch_bounds__(256) void ln_k(const float* __restrict__ Y,
                                            h16* __restrict__ Ynp,
                                            const float* __restrict__ sc,
                                            const float* __restrict__ bi) {
    const int lane = threadIdx.x & 31;
    const int wave = threadIdx.x >> 5;
    const long p   = (long)blockIdx.x * 8 + wave;   // NPIX divisible by 8

    const float4 x = *(const float4*)(Y + p * DDc + lane * 4);
    float s = x.x + x.y + x.z + x.w;
#pragma unroll
    for (int o = 16; o; o >>= 1) s += __shfl_xor(s, o, 32);
    const float mean = s * (1.0f / 128.0f);
    const float d0 = x.x - mean, d1 = x.y - mean, d2 = x.z - mean, d3 = x.w - mean;
    float v = d0 * d0 + d1 * d1 + d2 * d2 + d3 * d3;
#pragma unroll
    for (int o = 16; o; o >>= 1) v += __shfl_xor(v, o, 32);
    const float inv = rsqrtf(v * (1.0f / 128.0f) + 1e-5f);

    // padded destination index
    const int b = (int)(p / HWc);
    const int rem = (int)(p - (long)b * HWc);
    const int h = rem / Wc;
    const int w = rem - h * Wc;
    const long po = ((long)(b * HWp + (h + 1) * Wp + (w + 1))) * DDc;

    const int c = lane * 4;
    v4h o;
    o[0] = (h16)(d0 * inv * sc[c + 0] + bi[c + 0]);
    o[1] = (h16)(d1 * inv * sc[c + 1] + bi[c + 1]);
    o[2] = (h16)(d2 * inv * sc[c + 2] + bi[c + 2]);
    o[3] = (h16)(d3 * inv * sc[c + 3] + bi[c + 3]);
    *(v4h*)(Ynp + po + c) = o;
}

// ---------------- fused conv3x3(128->256) + GELU + 1x1(256->128) + residual ----------------
// Block = 128 threads = 4 waves; block owns 32 pixels (two 16-row M tiles sharing B).
// Conv GEMM: M=2x16, N=256 (wave w -> cols [64w,64w+64)), K=9*128 in steps of 32.
// All tap/chunk offsets are compile-time immediates off one padded base pointer.

__global__ __launch_bounds__(128) void conv_block_k(
    const h16*  __restrict__ Ynp,   // (B,28,28,128) f16, zero halo
    float*      __restrict__ Y,     // (B,26,26,128) f32 residual in/out
    const h16*  __restrict__ w3l,   // (9,128,256) f16
    const float* __restrict__ b3l,  // (256)
    const h16*  __restrict__ w1l,   // (256,128) f16
    const float* __restrict__ b1l)  // (128)
{
    __shared__ __align__(32) h16 gbuf[32 * CHc];   // 32x256 gelu tile (16 KB)

    const int lane = threadIdx.x & 31;
    const int wave = threadIdx.x >> 5;
    const int m    = lane & 15;
    const int hi   = lane >> 4;
    const int tile = blockIdx.x;

    // padded-corner base offsets (halves) for the lane's two pixel rows
    int base[2];
#pragma unroll
    for (int mt = 0; mt < 2; mt++) {
        const int pix = tile * 32 + mt * 16 + m;
        const int b   = pix / HWc;
        const int rem = pix - b * HWc;
        const int h   = rem / Wc;
        const int w   = rem - h * Wc;
        base[mt] = (b * HWp + h * Wp + w) * DDc;   // tap(0,0) corner in padded grid
    }
    const h16* ap0 = Ynp + base[0] + hi * 8;
    const h16* ap1 = Ynp + base[1] + hi * 8;
    const h16* bp  = w3l + (hi * 8) * CHc + wave * 64 + m; // tap/chunk offsets are immediates

    v8f acc[2][4] = {};

#pragma unroll
    for (int t = 0; t < 9; t++) {
        const int dy = t / 3, dx = t % 3;
        const int toff = (dy * Wp + dx) * DDc;     // compile-time constant
#pragma unroll
        for (int c0 = 0; c0 < DDc; c0 += 32) {
            v8h a00 = *(const v8h*)(ap0 + toff + c0);
            v8h a01 = *(const v8h*)(ap0 + toff + c0 + 16);
            v8h a10 = *(const v8h*)(ap1 + toff + c0);
            v8h a11 = *(const v8h*)(ap1 + toff + c0 + 16);
            v16h A0, A1;
#pragma unroll
            for (int j = 0; j < 8; j++) {
                A0[j] = a00[j]; A0[j + 8] = a01[j];
                A1[j] = a10[j]; A1[j + 8] = a11[j];
            }
            const int boff = (t * DDc + c0) * CHc; // compile-time constant
#pragma unroll
            for (int nt = 0; nt < 4; nt++) {
                v16h Bm;
#pragma unroll
                for (int j = 0; j < 8; j++) {
                    Bm[j]     = bp[boff + j * CHc + nt * 16];
                    Bm[j + 8] = bp[boff + (16 + j) * CHc + nt * 16];
                }
                acc[0][nt] = __builtin_amdgcn_wmma_f32_16x16x32_f16(
                    false, A0, false, Bm, (short)0, acc[0][nt], false, false);
                acc[1][nt] = __builtin_amdgcn_wmma_f32_16x16x32_f16(
                    false, A1, false, Bm, (short)0, acc[1][nt], false, false);
            }
        }
    }

    // epilogue 1: + b3, exact GELU, stage f16 tile to LDS
#pragma unroll
    for (int mt = 0; mt < 2; mt++) {
#pragma unroll
        for (int nt = 0; nt < 4; nt++) {
            const int ch = wave * 64 + nt * 16 + m;
            const float bv = b3l[ch];
#pragma unroll
            for (int r = 0; r < 8; r++) {
                const float x = acc[mt][nt][r] + bv;
                const float g = 0.5f * x * (1.0f + erff(x * 0.70710678118654752f));
                gbuf[(mt * 16 + r + 8 * hi) * CHc + ch] = (h16)g;
            }
        }
    }
    __syncthreads();

    // second GEMM: (32x256)x(256x128)
    v8f acc2[2][2] = {};
#pragma unroll
    for (int k0 = 0; k0 < CHc; k0 += 32) {
        const h16* lp0 = gbuf + m * CHc + k0 + hi * 8;
        const h16* lp1 = gbuf + (16 + m) * CHc + k0 + hi * 8;
        v8h a00 = *(const v8h*)(lp0);
        v8h a01 = *(const v8h*)(lp0 + 16);
        v8h a10 = *(const v8h*)(lp1);
        v8h a11 = *(const v8h*)(lp1 + 16);
        v16h A0, A1;
#pragma unroll
        for (int j = 0; j < 8; j++) {
            A0[j] = a00[j]; A0[j + 8] = a01[j];
            A1[j] = a10[j]; A1[j + 8] = a11[j];
        }
        const h16* wp = w1l + (k0 + hi * 8) * DDc + wave * 32 + m;
#pragma unroll
        for (int nt = 0; nt < 2; nt++) {
            v16h Bm;
#pragma unroll
            for (int j = 0; j < 8; j++) {
                Bm[j]     = wp[j * DDc + nt * 16];
                Bm[j + 8] = wp[(16 + j) * DDc + nt * 16];
            }
            acc2[0][nt] = __builtin_amdgcn_wmma_f32_16x16x32_f16(
                false, A0, false, Bm, (short)0, acc2[0][nt], false, false);
            acc2[1][nt] = __builtin_amdgcn_wmma_f32_16x16x32_f16(
                false, A1, false, Bm, (short)0, acc2[1][nt], false, false);
        }
    }

    // epilogue 2: + b1, residual add into Y
#pragma unroll
    for (int mt = 0; mt < 2; mt++) {
#pragma unroll
        for (int nt = 0; nt < 2; nt++) {
            const int d = wave * 32 + nt * 16 + m;
            const float bv = b1l[d];
#pragma unroll
            for (int r = 0; r < 8; r++) {
                const int mm = mt * 16 + r + 8 * hi;
                float* yp = Y + ((long)(tile * 32 + mm)) * DDc + d;
                *yp = *yp + acc2[mt][nt][r] + bv;
            }
        }
    }
}

// ---------------- launcher ----------------

extern "C" void kernel_launch(void* const* d_in, const int* in_sizes, int n_in,
                              void* d_out, int out_size, void* d_ws, size_t ws_size,
                              hipStream_t stream) {
    (void)in_sizes; (void)n_in; (void)out_size; (void)ws_size;

    const float* X      = (const float*)d_in[0];
    const int*   coords = (const int*)  d_in[1];
    const float* P      = (const float*)d_in[2];
    const float* ln_s   = (const float*)d_in[3];
    const float* ln_b   = (const float*)d_in[4];
    const float* w3     = (const float*)d_in[5];
    const float* b3     = (const float*)d_in[6];
    const float* w1     = (const float*)d_in[7];
    const float* b1     = (const float*)d_in[8];
    float* Z = (float*)d_out;

    // workspace carve-up
    char* ws = (char*)d_ws;
    float* Yg = (float*)ws;                                        // 88.6 MB
    size_t off = (size_t)NPIX * DDc * sizeof(float);
    h16* Ynp = (h16*)(ws + off); off += (size_t)Bc * HWp * DDc * sizeof(h16);      // 51.4 MB
    h16* w3h = (h16*)(ws + off); off += (size_t)NLc * 9 * DDc * CHc * sizeof(h16); // 2.25 MB
    h16* w1h = (h16*)(ws + off);                                                   // 0.25 MB

    const int n3 = NLc * 9 * DDc * CHc;   // 1,179,648
    const int n1 = NLc * CHc * DDc;       // 131,072
    cvt_f16_k<<<(n3 + 255) / 256, 256, 0, stream>>>(w3, w3h, n3);
    cvt_f16_k<<<(n1 + 255) / 256, 256, 0, stream>>>(w1, w1h, n1);

    // zero the whole padded Yn once: halo stays zero across layers
    const int n8 = (Bc * HWp * DDc) / 8;  // 3,211,264 v8h chunks
    zero_f16_k<<<(n8 + 255) / 256, 256, 0, stream>>>(Ynp, n8);

    const int nG = NPIX * DDc;            // 22,151,168
    init_grid_k<<<(nG + 255) / 256, 256, 0, stream>>>(Yg, P, nG);

    const int nS = Bc * Lc * DDc;         // 20,447,232
    scatter_k<<<(nS + 255) / 256, 256, 0, stream>>>(X, coords, Yg, nS);

    for (int l = 0; l < NLc; l++) {
        ln_k<<<NPIX / 8, 256, 0, stream>>>(Yg, Ynp, ln_s + l * DDc, ln_b + l * DDc);
        conv_block_k<<<NPIX / 32, 128, 0, stream>>>(
            Ynp, Yg,
            w3h + (size_t)l * 9 * DDc * CHc,
            b3  + (size_t)l * CHc,
            w1h + (size_t)l * CHc * DDc,
            b1  + (size_t)l * DDc);
    }

    gather_k<<<(nS + 255) / 256, 256, 0, stream>>>(Yg, coords, Z, nS);
}
